// GNS_39298950759109
// MI455X (gfx1250) — matrix operand order repaired and checked by reading, not compiled, not run on hardware
//
#include <hip/hip_runtime.h>
#include <hip/hip_bf16.h>

// ---------------- problem constants ----------------
#define GNS_N 10000
#define GNS_E 160000
#define GNS_L 10
#define ROWS 32          // rows per workgroup
#define AS   132         // fp32 act LDS stride (pad vs 128 to spread banks)
#define HS   136         // bf16 hidden LDS stride (pad, keeps 16B alignment)
#define DPADH 8          // bf16 input-tile row pad (elements)
#define LN_EPS 1e-5f

typedef __bf16 v16bf __attribute__((ext_vector_type(16)));
typedef float  v8f   __attribute__((ext_vector_type(8)));

union Frag16 { uint4 q[2]; v16bf v; };

__device__ __forceinline__ unsigned short f2bf(float f) {
  unsigned u = __float_as_uint(f);
  u += 0x7fffu + ((u >> 16) & 1u);      // round-to-nearest-even
  return (unsigned short)(u >> 16);
}

__device__ __forceinline__ float silu_f(float x) { return x / (1.0f + __expf(-x)); }

// input segment descriptor: contributes `cols` columns, source row stride == cols
struct Seg { const void* ptr; const int* idx; int cols; int isf32; };

// ---- WMMA fragment loaders -------------------------------------------------
// A (16x32 bf16, 16-bit A layout): lane m=lane&15, kh=lane>>4;
// VGPR0..3 hold K = kh*8 + 0..7 (pairs), VGPR4..7 hold K = 16 + kh*8 + 0..7.
__device__ __forceinline__ v16bf load_a(const unsigned short* lds, int rbase,
                                        int strideHalf, int kt, int lane) {
  int m = lane & 15, kh = lane >> 4;
  const unsigned short* p = lds + (rbase + m) * strideHalf + kt * 32 + kh * 8;
  Frag16 f;
  f.q[0] = *(const uint4*)p;        // K kh*8 .. kh*8+7
  f.q[1] = *(const uint4*)(p + 16); // K 16+kh*8 .. 16+kh*8+7
  return f.v;
}

// B (32x16 bf16) pre-swizzled at pack time: tile (nt,kt) -> 32 lanes x 16 bf16,
// lane holds n = nt*16 + (lane&15), k = kt*32 + (j>>3)*16 + (lane>>4)*8 + (j&7), j=0..15.
__device__ __forceinline__ v16bf load_b(const unsigned short* wp, int nt, int kt,
                                        int nkt, int lane) {
  const unsigned short* p = wp + ((size_t)(nt * nkt + kt) * 32 + lane) * 16;
  Frag16 f;
  f.q[0] = *(const uint4*)p;
  f.q[1] = *(const uint4*)(p + 8);
  return f.v;
}

__device__ __forceinline__ void gemm2x(const unsigned short* Alds, int strideHalf,
                                       int nkt, const unsigned short* Wp,
                                       int w, int lane, v8f acc[2]) {
  for (int kt = 0; kt < nkt; ++kt) {
    v16bf b = load_b(Wp, w, kt, nkt, lane);
#pragma unroll
    for (int rt = 0; rt < 2; ++rt) {
      v16bf a = load_a(Alds, rt * 16, strideHalf, kt, lane);
      acc[rt] = __builtin_amdgcn_wmma_f32_16x16x32_bf16(
          false, a, false, b, (short)0, acc[rt], false, false);
    }
  }
}

// ---- weight pack: fp32 [K x 128] row-major -> bf16 WMMA-B-swizzled tiles ----
__global__ __launch_bounds__(256)
void pack_w_kernel(const float* __restrict__ W, unsigned short* __restrict__ Wp,
                   int K, int nkt) {
  int idx = blockIdx.x * 256 + threadIdx.x;
  int total = 8 * nkt * 512;
  if (idx >= total) return;
  int j    = idx & 15;
  int lane = (idx >> 4) & 31;
  int t    = idx >> 9;
  int kt   = t % nkt;
  int nt   = t / nkt;
  int k = kt * 32 + (j >> 3) * 16 + (lane >> 4) * 8 + (j & 7);
  int n = nt * 16 + (lane & 15);
  float v = (k < K) ? W[(size_t)k * 128 + n] : 0.0f;
  Wp[idx] = f2bf(v);
}

__global__ __launch_bounds__(256)
void zero_f_kernel(float* __restrict__ p, int n) {
  int i = blockIdx.x * 256 + threadIdx.x;
  if (i < n) p[i] = 0.0f;
}

// segment_sum(e, dst) via native fp32 global atomics
__global__ __launch_bounds__(256)
void scatter_add_kernel(const float* __restrict__ e, const int* __restrict__ dst,
                        float* __restrict__ agg, int nE) {
  int i = blockIdx.x * 256 + threadIdx.x;
  if (i < nE * 128) {
    int eid = i >> 7, c = i & 127;
    __hip_atomic_fetch_add(&agg[(size_t)dst[eid] * 128 + c], e[i],
                           __ATOMIC_RELAXED, __HIP_MEMORY_SCOPE_AGENT);
  }
}

// ---- fused 3-layer MLP (+optional LayerNorms, residual, bf16 mirror) -------
// mode 0: y = LN(silu(.W1+b1)) -> LN(silu(.W2+b2)) -> .W3+b3 (+res), writes out+outbf
// mode 1: decoder: silu, silu, then scalar 128->3 with decW3/decb3 into out
__global__ __launch_bounds__(256)
void mlp3_wmma(Seg s0, Seg s1, Seg s2, int din, int nrows,
               const unsigned short* __restrict__ W1p,
               const unsigned short* __restrict__ W2p,
               const unsigned short* __restrict__ W3p,
               const float* __restrict__ b1, const float* __restrict__ g1,
               const float* __restrict__ be1,
               const float* __restrict__ b2, const float* __restrict__ g2,
               const float* __restrict__ be2,
               const float* __restrict__ b3,
               float* __restrict__ out, unsigned short* __restrict__ outbf,
               int res, int mode,
               const float* __restrict__ decW3, const float* __restrict__ decb3)
{
  __shared__ __align__(16) unsigned short in_tile[ROWS * (384 + DPADH)];
  __shared__ __align__(16) float          act[ROWS * AS];
  __shared__ __align__(16) unsigned short hbuf[ROWS * HS];
  __shared__ float red[ROWS * 8 * 2];
  __shared__ float mv[ROWS * 2];

  const int tid  = threadIdx.x;
  const int lane = tid & 31;
  const int w    = tid >> 5;      // 8 waves -> 8 column tiles of 16
  const int n    = lane & 15;
  const int mh   = lane >> 4;
  const int rowbase = blockIdx.x * ROWS;
  const int S = din + DPADH;      // bf16 input-tile stride (S % 8 == 0)

  // ---- assemble input tile (concat segments, fp32->bf16, zero pad) ----
  const int o1 = s0.cols, o2 = o1 + s1.cols, o3 = o2 + s2.cols;
  for (int t = tid; t < ROWS * S; t += 256) {
    int r = t / S, c = t - r * S;
    int gr = rowbase + r;
    unsigned short hv = 0;
    if (gr < nrows && c < din) {
      const Seg* sg = nullptr; int cc = 0;
      if (c < o1)      { sg = &s0; cc = c; }
      else if (c < o2) { sg = &s1; cc = c - o1; }
      else if (c < o3) { sg = &s2; cc = c - o2; }
      if (sg && sg->ptr) {
        int sr = sg->idx ? sg->idx[gr] : gr;
        if (sg->isf32)
          hv = f2bf(((const float*)sg->ptr)[(size_t)sr * sg->cols + cc]);
        else
          hv = ((const unsigned short*)sg->ptr)[(size_t)sr * sg->cols + cc];
      }
    }
    in_tile[t] = hv;
  }
  __syncthreads();

  v8f acc[2];
  const v8f zacc = {0.f, 0.f, 0.f, 0.f, 0.f, 0.f, 0.f, 0.f};

  // C/D layout: row = rt*16 + mh*8 + vg, col = w*16 + n
  auto store_act = [&](const float* bias_v, bool dosilu) {
    float bias = bias_v[w * 16 + n];
#pragma unroll
    for (int rt = 0; rt < 2; ++rt)
#pragma unroll
      for (int vg = 0; vg < 8; ++vg) {
        float x = acc[rt][vg] + bias;
        act[(rt * 16 + mh * 8 + vg) * AS + w * 16 + n] = dosilu ? silu_f(x) : x;
      }
  };

  // LayerNorm over 128 cols of act -> bf16 hbuf. 8 threads per row.
  auto ln_phase = [&](const float* g, const float* be) {
    int r = tid >> 3, p = tid & 7;
    float s = 0.f, ss = 0.f;
#pragma unroll
    for (int c = 0; c < 16; ++c) {
      float v = act[r * AS + p * 16 + c];
      s += v; ss += v * v;
    }
    red[(r * 8 + p) * 2 + 0] = s;
    red[(r * 8 + p) * 2 + 1] = ss;
    __syncthreads();
    if (tid < ROWS) {
      float ts = 0.f, tss = 0.f;
#pragma unroll
      for (int q = 0; q < 8; ++q) {
        ts  += red[(tid * 8 + q) * 2 + 0];
        tss += red[(tid * 8 + q) * 2 + 1];
      }
      float m = ts * (1.0f / 128.0f);
      float var = tss * (1.0f / 128.0f) - m * m;
      mv[tid * 2 + 0] = m;
      mv[tid * 2 + 1] = rsqrtf(var + LN_EPS);
    }
    __syncthreads();
    float m = mv[r * 2], rs = mv[r * 2 + 1];
#pragma unroll
    for (int c = 0; c < 16; ++c) {
      int col = p * 16 + c;
      float x = (act[r * AS + col] - m) * rs * g[col] + be[col];
      hbuf[r * HS + col] = f2bf(x);
    }
  };

  auto copy_act_to_hbuf = [&]() {
    for (int t = tid; t < ROWS * 128; t += 256) {
      int r = t >> 7, c = t & 127;
      hbuf[r * HS + c] = f2bf(act[r * AS + c]);
    }
  };

  // ===== stage 1: din -> 128 =====
  acc[0] = zacc; acc[1] = zacc;
  gemm2x(in_tile, S, din >> 5, W1p, w, lane, acc);
  store_act(b1, true);
  __syncthreads();
  if (mode == 0) ln_phase(g1, be1); else copy_act_to_hbuf();
  __syncthreads();

  // ===== stage 2: 128 -> 128 =====
  acc[0] = zacc; acc[1] = zacc;
  gemm2x(hbuf, HS, 4, W2p, w, lane, acc);
  store_act(b2, true);
  __syncthreads();

  if (mode == 0) {
    ln_phase(g2, be2);
    __syncthreads();

    // ===== stage 3: 128 -> 128 (+bias, +residual), fp32 + bf16 stores =====
    acc[0] = zacc; acc[1] = zacc;
    gemm2x(hbuf, HS, 4, W3p, w, lane, acc);
    store_act(b3, false);
    __syncthreads();
    for (int t = tid; t < ROWS * 128; t += 256) {
      int r = t >> 7, c = t & 127;
      int gr = rowbase + r;
      if (gr < nrows) {
        size_t gi = (size_t)gr * 128 + c;
        float v = act[r * AS + c];
        if (res) v += out[gi];
        out[gi] = v;
        outbf[gi] = f2bf(v);
      }
    }
  } else {
    // decoder tail: act holds silu(h2); scalar 128 -> 3
    if (tid < ROWS * 3) {
      int r = tid / 3, cc = tid - r * 3;
      int gr = rowbase + r;
      if (gr < nrows) {
        float y = decb3[cc];
        for (int k = 0; k < 128; ++k)
          y += act[r * AS + k] * decW3[k * 3 + cc];
        out[(size_t)gr * 3 + cc] = y;
      }
    }
  }
}

// ---------------- host orchestration ----------------
extern "C" void kernel_launch(void* const* d_in, const int* in_sizes, int n_in,
                              void* d_out, int out_size, void* d_ws, size_t ws_size,
                              hipStream_t stream) {
  (void)in_sizes; (void)n_in; (void)out_size; (void)ws_size;
  const int N = GNS_N, E = GNS_E;
  const float* x  = (const float*)d_in[0];
  const int*   ei = (const int*)d_in[1];
  const float* ea = (const float*)d_in[2];
  const int* ei_src = ei;
  const int* ei_dst = ei + E;
  auto F = [&](int i) { return (const float*)d_in[i]; };

  // workspace layout
  char* base = (char*)d_ws;
  size_t off = 0;
  auto alloc = [&](size_t bytes) -> void* {
    void* p = base + off;
    off = (off + bytes + 255) & ~(size_t)255;
    return p;
  };
  float* h_f = (float*)alloc((size_t)N * 128 * 4);
  float* e_f = (float*)alloc((size_t)E * 128 * 4);
  float* agg = (float*)alloc((size_t)N * 128 * 4);
  unsigned short* h_bf = (unsigned short*)alloc((size_t)N * 128 * 2);
  unsigned short* e_bf = (unsigned short*)alloc((size_t)E * 128 * 2);

  auto packW = [&](const float* src, int K) -> unsigned short* {
    int nkt = ((K + 31) & ~31) >> 5;
    int total = 8 * nkt * 512;
    unsigned short* dst = (unsigned short*)alloc((size_t)total * 2);
    pack_w_kernel<<<(total + 255) / 256, 256, 0, stream>>>(src, dst, K, nkt);
    return dst;
  };

  // pack all GEMM weights to WMMA-swizzled bf16
  unsigned short* ncW1 = packW(F(3), 30);
  unsigned short* ncW2 = packW(F(7), 128);
  unsigned short* ncW3 = packW(F(11), 128);
  unsigned short* ecW1 = packW(F(13), 4);
  unsigned short* ecW2 = packW(F(17), 128);
  unsigned short* ecW3 = packW(F(21), 128);
  unsigned short *leW1[GNS_L], *leW2[GNS_L], *leW3[GNS_L];
  unsigned short *lnW1[GNS_L], *lnW2[GNS_L], *lnW3[GNS_L];
  for (int l = 0; l < GNS_L; ++l) {
    leW1[l] = packW(F(23) + (size_t)l * 384 * 128, 384);
    leW2[l] = packW(F(27) + (size_t)l * 128 * 128, 128);
    leW3[l] = packW(F(31) + (size_t)l * 128 * 128, 128);
    lnW1[l] = packW(F(33) + (size_t)l * 256 * 128, 256);
    lnW2[l] = packW(F(37) + (size_t)l * 128 * 128, 128);
    lnW3[l] = packW(F(41) + (size_t)l * 128 * 128, 128);
  }
  unsigned short* dW1 = packW(F(43), 128);
  unsigned short* dW2 = packW(F(45), 128);

  const int gridN = (N + ROWS - 1) / ROWS;
  const int gridE = (E + ROWS - 1) / ROWS;
  Seg z{nullptr, nullptr, 0, 0};

  // node encoder: h = mlp_ln(x)  (din 30 -> padded 32)
  {
    Seg sx{x, nullptr, 30, 1};
    mlp3_wmma<<<gridN, 256, 0, stream>>>(sx, z, z, 32, N,
        ncW1, ncW2, ncW3, F(4), F(5), F(6), F(8), F(9), F(10), F(12),
        h_f, h_bf, 0, 0, nullptr, nullptr);
  }
  // edge encoder: e = mlp_ln(edge_attr)  (din 4 -> padded 32)
  {
    Seg se{ea, nullptr, 4, 1};
    mlp3_wmma<<<gridE, 256, 0, stream>>>(se, z, z, 32, E,
        ecW1, ecW2, ecW3, F(14), F(15), F(16), F(18), F(19), F(20), F(22),
        e_f, e_bf, 0, 0, nullptr, nullptr);
  }

  for (int l = 0; l < GNS_L; ++l) {
    const float* eb1 = F(24) + l * 128; const float* eg1 = F(25) + l * 128;
    const float* ee1 = F(26) + l * 128; const float* eb2 = F(28) + l * 128;
    const float* eg2 = F(29) + l * 128; const float* ee2 = F(30) + l * 128;
    const float* eb3 = F(32) + l * 128;
    const float* nb1 = F(34) + l * 128; const float* ng1 = F(35) + l * 128;
    const float* ne1 = F(36) + l * 128; const float* nb2 = F(38) + l * 128;
    const float* ng2 = F(39) + l * 128; const float* ne2 = F(40) + l * 128;
    const float* nb3 = F(42) + l * 128;

    // edge update: e += mlp_ln([h[src], h[dst], e])
    Seg g0{h_bf, ei_src, 128, 0};
    Seg g1{h_bf, ei_dst, 128, 0};
    Seg g2{e_bf, nullptr, 128, 0};
    mlp3_wmma<<<gridE, 256, 0, stream>>>(g0, g1, g2, 384, E,
        leW1[l], leW2[l], leW3[l], eb1, eg1, ee1, eb2, eg2, ee2, eb3,
        e_f, e_bf, 1, 0, nullptr, nullptr);

    // agg = segment_sum(e, dst)
    zero_f_kernel<<<(N * 128 + 255) / 256, 256, 0, stream>>>(agg, N * 128);
    scatter_add_kernel<<<(E * 128 + 255) / 256, 256, 0, stream>>>(e_f, ei_dst, agg, E);

    // node update: h += mlp_ln([h, agg])
    Seg n0{h_bf, nullptr, 128, 0};
    Seg n1{agg,  nullptr, 128, 1};
    mlp3_wmma<<<gridN, 256, 0, stream>>>(n0, n1, z, 256, N,
        lnW1[l], lnW2[l], lnW3[l], nb1, ng1, ne1, nb2, ng2, ne2, nb3,
        h_f, h_bf, 1, 0, nullptr, nullptr);
  }

  // decoder: out = mlp(h)  (silu, silu, scalar 128->3)
  {
    Seg d0{h_bf, nullptr, 128, 0};
    mlp3_wmma<<<gridN, 256, 0, stream>>>(d0, z, z, 128, N,
        dW1, dW2, nullptr, F(44), nullptr, nullptr, F(46), nullptr, nullptr,
        nullptr, (float*)d_out, nullptr, 0, 1, F(47), F(48));
  }
}
